// T5Attention_86234353369205
// MI455X (gfx1250) — compile-verified, matrix operand order checked by reading, not run on hardware
//
#include <hip/hip_runtime.h>
#include <stdint.h>

// ---------------- problem constants (match reference) ----------------
#define B_   2
#define S_   2048
#define H_   8
#define DKV  64
#define DM   512      // d_model = H_*DKV

typedef __attribute__((ext_vector_type(16))) __bf16       v16bf;
typedef __attribute__((ext_vector_type(8)))  float        v8f;
typedef __attribute__((ext_vector_type(4)))  unsigned int u32x4;
typedef __attribute__((ext_vector_type(4)))  int          v4i;

// address-space-qualified int4 pointee types for the async-LDS builtin
typedef __attribute__((address_space(1))) v4i gv4i;   // global
typedef __attribute__((address_space(3))) v4i lv4i;   // LDS

union FragU { v16bf v; u32x4 q[2]; };

#if __has_builtin(__builtin_amdgcn_global_load_async_to_lds_b128)
#define ASYNC_LDS 1
#endif

__device__ __forceinline__ uint16_t f2bf(float x) {
  union { float f; uint32_t u; } c; c.f = x;
  uint32_t r = c.u + 0x7FFFu + ((c.u >> 16) & 1u);   // round-to-nearest-even
  return (uint16_t)(r >> 16);
}

// Copy 16 bytes global -> LDS. Async (GLOBAL_LOAD_ASYNC_TO_LDS_B128, ASYNCcnt)
// when the toolchain exposes it, else synchronous load + ds_store.
__device__ __forceinline__ void cp16(uint16_t* ldst, const uint16_t* gsrc) {
#ifdef ASYNC_LDS
  __builtin_amdgcn_global_load_async_to_lds_b128(
      (gv4i*)(uintptr_t)gsrc,                 // generic global ptr value == AS1 value
      (lv4i*)(uint32_t)(uintptr_t)ldst,       // low 32 bits of flat LDS addr == LDS offset
      0, 0);
#else
  *(u32x4*)ldst = *(const u32x4*)gsrc;
#endif
}

__device__ __forceinline__ void wait_async_lds() {
#ifdef ASYNC_LDS
#if __has_builtin(__builtin_amdgcn_s_wait_asynccnt)
  __builtin_amdgcn_s_wait_asynccnt(0);
#else
  asm volatile("s_wait_asynccnt 0" ::: "memory");
#endif
#endif
}

// Load one 16x32 bf16 WMMA operand fragment (A-layout; identical pattern for
// B when the B matrix is stored N-major, i.e. one column == contiguous row).
// lane<16 : row=lane,    K = {0..7, 16..23}
// lane>=16: row=lane-16, K = {8..15, 24..31}
// Works for global or LDS-derived pointers (addrspace inferred after inline).
__device__ __forceinline__ v16bf load_frag(const uint16_t* tile, int ld, int kbase,
                                           unsigned lane) {
  const uint16_t* rp = tile + (size_t)(lane & 15u) * (size_t)ld + kbase
                            + ((lane & 16u) ? 8 : 0);
  FragU f;
  f.q[0] = *(const u32x4*)(rp);
  f.q[1] = *(const u32x4*)(rp + 16);
  return f.v;
}

#define WMMA_BF16(A, Bf, C) \
  __builtin_amdgcn_wmma_f32_16x16x32_bf16(false, (A), false, (Bf), (short)0, (C), false, false)

// ---------------- prep kernels ----------------
__global__ void cvt_hidden_kernel(const float* __restrict__ src, uint16_t* __restrict__ dst) {
  size_t i = (size_t)blockIdx.x * 256 + threadIdx.x;
  dst[i] = f2bf(src[i]);
}

// weights arrive [K=512][N=512]; store bf16 transposed [N][K] so B-fragment
// columns are contiguous.
__global__ void cvt_weightT_kernel(const float* __restrict__ w0, const float* __restrict__ w1,
                                   const float* __restrict__ w2, const float* __restrict__ w3,
                                   uint16_t* __restrict__ dst) {
  int i = blockIdx.x * 256 + threadIdx.x;          // i = n*512 + k
  const float* src = (blockIdx.y == 0) ? w0 : (blockIdx.y == 1) ? w1
                   : (blockIdx.y == 2) ? w2 : w3;
  int n = i >> 9, k = i & 511;
  dst[(size_t)blockIdx.y * (DM * DM) + i] = f2bf(src[(size_t)k * DM + n]);
}

// precompute bias[rel + (S-1)][h] once.
__global__ void bias_table_kernel(const float* __restrict__ table, float* __restrict__ out) {
  int idx = blockIdx.x * 256 + threadIdx.x;
  if (idx >= (2 * S_ - 1) * H_) return;
  int rel = idx / H_ - (S_ - 1);                   // k - q
  int h = idx % H_;
  int bucket = (rel > 0) ? 16 : 0;                 // num_buckets/2
  int rp = (rel < 0) ? -rel : rel;
  int bv;
  if (rp < 8) {                                    // max_exact = 8
    bv = rp;
  } else {
    int lg = 8 + (int)(__logf((float)rp * 0.125f) * (8.0f / __logf(16.0f)));
    bv = lg < 15 ? lg : 15;
  }
  out[idx] = table[(bucket + bv) * H_ + h];
}

// ---------------- QKV projection GEMM (bf16 WMMA, LDS-staged weights) ----------------
// grid = ( (B*S)/128, 3*H ); block = 256 (8 waves). Each wave: 16x64 tile.
// All 8 waves share the same 64-wide weight slab -> stage 64x32 K-tiles in LDS,
// double-buffered, async.
__global__ __launch_bounds__(256)
void qkv_gemm_kernel(const uint16_t* __restrict__ hid, const uint16_t* __restrict__ wT,
                     uint16_t* __restrict__ qs, uint16_t* __restrict__ ks,
                     uint16_t* __restrict__ vs) {
  __shared__ uint16_t wl[2][64][40];               // 64 N x 32 K (+pad) x2 buf = 10 KB
  const unsigned lane = threadIdx.x & 31u, wave = threadIdx.x >> 5;
  const int tid = threadIdx.x;
  const int mat = blockIdx.y >> 3, h = blockIdx.y & 7;
  const int m0 = blockIdx.x * 128 + wave * 16;
  const uint16_t* w = wT + (size_t)mat * DM * DM + (size_t)h * 64 * DM;
  const uint16_t* a_tile = hid + (size_t)m0 * DM;

  v8f acc[4];
#pragma unroll
  for (int ns = 0; ns < 4; ++ns)
#pragma unroll
    for (int i = 0; i < 8; ++i) acc[ns][i] = 0.f;

  // stage k-tile 0: 64 rows x 4 chunks(16B) = 256 chunks, one per thread
  {
    int row = tid >> 2, ch = tid & 3;
    cp16(&wl[0][row][ch * 8], w + (size_t)row * DM + ch * 8);
  }
  int cur = 0;
  for (int k = 0; k < DM; k += 32) {
    wait_async_lds();
    __syncthreads();                                // tile `cur` ready; buf cur^1 free
    if (k + 32 < DM) {
      int row = tid >> 2, ch = tid & 3;
      cp16(&wl[cur ^ 1][row][ch * 8], w + (size_t)row * DM + (k + 32) + ch * 8);
    }
    v16bf a = load_frag(a_tile, DM, k, lane);
#pragma unroll
    for (int ns = 0; ns < 4; ++ns) {
      v16bf b = load_frag(&wl[cur][ns * 16][0], 40, 0, lane);
      acc[ns] = WMMA_BF16(a, b, acc[ns]);
    }
    cur ^= 1;
  }

  const int bi = m0 >> 11, s0 = m0 & (S_ - 1);
  const int hi = (lane >> 4) & 1, ln = lane & 15;
  const size_t bh = (size_t)(bi * H_ + h);
#pragma unroll
  for (int ns = 0; ns < 4; ++ns)
#pragma unroll
    for (int i = 0; i < 8; ++i) {
      int s = s0 + i + 8 * hi, d = ns * 16 + ln;
      uint16_t val = f2bf(acc[ns][i]);
      if (mat == 0)      qs[(bh * S_ + s) * DKV + d] = val;
      else if (mat == 1) ks[(bh * S_ + s) * DKV + d] = val;
      else               vs[(bh * DKV + d) * S_ + s] = val;   // V transposed [d][S]
    }
}

// ---------------- fused flash attention (bf16 WMMA, online softmax) ----------------
// grid = (S/128, B*H); block = 256 (8 waves, each owns 16 q-rows).
// K-tile (64x64) and V^T-tile (64x64) staged cooperatively into LDS, async,
// double-buffered; one barrier per k-tile.
__global__ __launch_bounds__(256)
void flash_attn_kernel(const uint16_t* __restrict__ qs, const uint16_t* __restrict__ ks,
                       const uint16_t* __restrict__ vs, const float* __restrict__ bias,
                       uint16_t* __restrict__ attn) {
  __shared__ uint16_t ktl[2][64][72];   // 18.4 KB
  __shared__ uint16_t vtl[2][64][72];   // 18.4 KB
  __shared__ uint16_t plds[8][16][72];  // per-wave P tile, 18.4 KB
  const unsigned lane = threadIdx.x & 31u, wave = threadIdx.x >> 5;
  const int tid = threadIdx.x;
  const int bh = blockIdx.y, b = bh >> 3, h = bh & 7;
  const int q0 = blockIdx.x * 128 + wave * 16;
  const uint16_t* qb = qs + (size_t)bh * S_ * DKV + (size_t)q0 * DKV;
  const uint16_t* kb = ks + (size_t)bh * S_ * DKV;
  const uint16_t* vb = vs + (size_t)bh * DKV * S_;        // [d][S]
  const float* biasr = bias + (size_t)(S_ - 1) * H_ + h;  // index by rel*H
  const int hi = (lane >> 4) & 1, ln = lane & 15;

  // Q fragments are loop-invariant: load once (16 VGPRs held live).
  v16bf qa0 = load_frag(qb, DKV, 0, lane);
  v16bf qa1 = load_frag(qb, DKV, 32, lane);

  float mrun[8], lrun[8];
  v8f o[4];
#pragma unroll
  for (int i = 0; i < 8; ++i) { mrun[i] = -1e30f; lrun[i] = 0.f; }
#pragma unroll
  for (int ns = 0; ns < 4; ++ns)
#pragma unroll
    for (int i = 0; i < 8; ++i) o[ns][i] = 0.f;

  // stage tile 0: K 512 chunks + V 512 chunks; 256 threads x (2+2)
  {
#pragma unroll
    for (int r = 0; r < 2; ++r) {
      int idx = tid + r * 256;
      int row = idx >> 3, ch = idx & 7;
      cp16(&ktl[0][row][ch * 8], kb + (size_t)row * DKV + ch * 8);
      cp16(&vtl[0][row][ch * 8], vb + (size_t)row * S_ + ch * 8);
    }
  }
  int cur = 0;
  for (int k0 = 0; k0 < S_; k0 += 64) {
    wait_async_lds();
    __syncthreads();                       // tiles[cur] ready; tiles[cur^1] free
    if (k0 + 64 < S_) {
      int nk0 = k0 + 64;
#pragma unroll
      for (int r = 0; r < 2; ++r) {
        int idx = tid + r * 256;
        int row = idx >> 3, ch = idx & 7;
        cp16(&ktl[cur ^ 1][row][ch * 8], kb + (size_t)(nk0 + row) * DKV + ch * 8);
        cp16(&vtl[cur ^ 1][row][ch * 8], vb + (size_t)row * S_ + nk0 + ch * 8);
      }
    }
    // ---- S = Q K^T for a 16x64 tile (B fragments from LDS) ----
    v8f sc[4];
#pragma unroll
    for (int ns = 0; ns < 4; ++ns)
#pragma unroll
      for (int i = 0; i < 8; ++i) sc[ns][i] = 0.f;
#pragma unroll
    for (int ns = 0; ns < 4; ++ns) {
      v16bf b0 = load_frag(&ktl[cur][ns * 16][0], 72, 0, lane);
      sc[ns] = WMMA_BF16(qa0, b0, sc[ns]);
      v16bf b1 = load_frag(&ktl[cur][ns * 16][0], 72, 32, lane);
      sc[ns] = WMMA_BF16(qa1, b1, sc[ns]);
    }
    // ---- bias + scale + online softmax (row = i + 8*hi, col = ns*16 + ln) ----
#pragma unroll
    for (int i = 0; i < 8; ++i) {
      int qi = q0 + i + 8 * hi;
      float rmax = -1e30f;
#pragma unroll
      for (int ns = 0; ns < 4; ++ns) {
        int kid = k0 + ns * 16 + ln;
        float lg = (sc[ns][i] + biasr[(kid - qi) * H_]) * 0.125f;  // 1/sqrt(64)
        sc[ns][i] = lg;
        rmax = fmaxf(rmax, lg);
      }
      rmax = fmaxf(rmax, __shfl_xor(rmax, 1, 32));
      rmax = fmaxf(rmax, __shfl_xor(rmax, 2, 32));
      rmax = fmaxf(rmax, __shfl_xor(rmax, 4, 32));
      rmax = fmaxf(rmax, __shfl_xor(rmax, 8, 32));
      float mnew  = fmaxf(mrun[i], rmax);
      float alpha = __expf(mrun[i] - mnew);
      mrun[i] = mnew;
      float rs = 0.f;
#pragma unroll
      for (int ns = 0; ns < 4; ++ns) {
        float p = __expf(sc[ns][i] - mnew);
        sc[ns][i] = p;
        rs += p;
      }
      rs += __shfl_xor(rs, 1, 32);
      rs += __shfl_xor(rs, 2, 32);
      rs += __shfl_xor(rs, 4, 32);
      rs += __shfl_xor(rs, 8, 32);
      lrun[i] = lrun[i] * alpha + rs;
#pragma unroll
      for (int ns = 0; ns < 4; ++ns) o[ns][i] *= alpha;   // rescale O before +=
    }
    // ---- C-layout P -> LDS -> A-layout (per-wave region; DS is in-order) ----
#pragma unroll
    for (int ns = 0; ns < 4; ++ns)
#pragma unroll
      for (int i = 0; i < 8; ++i)
        plds[wave][i + 8 * hi][ns * 16 + ln] = f2bf(sc[ns][i]);
    asm volatile("s_wait_dscnt 0" ::: "memory");
    // ---- O += P V (both operands from LDS) ----
#pragma unroll
    for (int ksub = 0; ksub < 2; ++ksub) {
      const uint16_t* pr = &plds[wave][ln][ksub * 32 + (hi ? 8 : 0)];
      FragU fa;
      fa.q[0] = *(const u32x4*)(pr);
      fa.q[1] = *(const u32x4*)(pr + 16);
#pragma unroll
      for (int ns = 0; ns < 4; ++ns) {
        v16bf bf = load_frag(&vtl[cur][ns * 16][0], 72, ksub * 32, lane);
        o[ns] = WMMA_BF16(fa.v, bf, o[ns]);
      }
    }
    cur ^= 1;
  }
  // ---- normalize and store heads-merged [B*S][512] bf16 ----
#pragma unroll
  for (int i = 0; i < 8; ++i) {
    float inv = 1.0f / lrun[i];
    int s = q0 + i + 8 * hi;
#pragma unroll
    for (int ns = 0; ns < 4; ++ns)
      attn[((size_t)(b * S_) + s) * DM + h * 64 + ns * 16 + ln] = f2bf(o[ns][i] * inv);
  }
}

// ---------------- output projection GEMM (bf16 WMMA -> f32 out) ----------------
__global__ __launch_bounds__(256)
void out_gemm_kernel(const uint16_t* __restrict__ attn, const uint16_t* __restrict__ woT,
                     float* __restrict__ out) {
  __shared__ uint16_t wl[2][64][40];
  const unsigned lane = threadIdx.x & 31u, wave = threadIdx.x >> 5;
  const int tid = threadIdx.x;
  const int m0 = blockIdx.x * 128 + wave * 16;
  const int n0 = blockIdx.y * 64;
  const uint16_t* w = woT + (size_t)n0 * DM;
  const uint16_t* a_tile = attn + (size_t)m0 * DM;

  v8f acc[4];
#pragma unroll
  for (int ns = 0; ns < 4; ++ns)
#pragma unroll
    for (int i = 0; i < 8; ++i) acc[ns][i] = 0.f;

  {
    int row = tid >> 2, ch = tid & 3;
    cp16(&wl[0][row][ch * 8], w + (size_t)row * DM + ch * 8);
  }
  int cur = 0;
  for (int k = 0; k < DM; k += 32) {
    wait_async_lds();
    __syncthreads();
    if (k + 32 < DM) {
      int row = tid >> 2, ch = tid & 3;
      cp16(&wl[cur ^ 1][row][ch * 8], w + (size_t)row * DM + (k + 32) + ch * 8);
    }
    v16bf a = load_frag(a_tile, DM, k, lane);
#pragma unroll
    for (int ns = 0; ns < 4; ++ns) {
      v16bf b = load_frag(&wl[cur][ns * 16][0], 40, 0, lane);
      acc[ns] = WMMA_BF16(a, b, acc[ns]);
    }
    cur ^= 1;
  }
  const int hi = (lane >> 4) & 1, ln = lane & 15;
#pragma unroll
  for (int ns = 0; ns < 4; ++ns)
#pragma unroll
    for (int i = 0; i < 8; ++i)
      out[(size_t)(m0 + i + 8 * hi) * DM + n0 + ns * 16 + ln] = acc[ns][i];
}

// ---------------- launcher ----------------
extern "C" void kernel_launch(void* const* d_in, const int* in_sizes, int n_in,
                              void* d_out, int out_size, void* d_ws, size_t ws_size,
                              hipStream_t stream) {
  const float* hidden = (const float*)d_in[0];
  const float* wq     = (const float*)d_in[1];
  const float* wk     = (const float*)d_in[2];
  const float* wv     = (const float*)d_in[3];
  const float* wo     = (const float*)d_in[4];
  const float* table  = (const float*)d_in[5];
  float* out = (float*)d_out;

  // workspace layout (~23.2 MB total)
  char* ws = (char*)d_ws;
  constexpr size_t sz_hid  = (size_t)B_ * S_ * DM * 2;       // 4 MB bf16 hidden
  constexpr size_t sz_w    = (size_t)DM * DM * 2 * 4;        // 2 MB bf16 wT x4
  constexpr size_t sz_qkv  = (size_t)B_ * H_ * S_ * DKV * 2; // 4 MB each
  uint16_t* hid_bf = (uint16_t*)(ws);
  uint16_t* w_bf   = (uint16_t*)(ws + sz_hid);
  uint16_t* qbuf   = (uint16_t*)(ws + sz_hid + sz_w);
  uint16_t* kbuf   = (uint16_t*)(ws + sz_hid + sz_w + sz_qkv);
  uint16_t* vbuf   = (uint16_t*)(ws + sz_hid + sz_w + 2 * sz_qkv);
  uint16_t* abuf   = (uint16_t*)(ws + sz_hid + sz_w + 3 * sz_qkv);
  float*    biasbf = (float*)   (ws + sz_hid + sz_w + 3 * sz_qkv + sz_hid);

  cvt_hidden_kernel<<<(B_ * S_ * DM) / 256, 256, 0, stream>>>(hidden, hid_bf);
  cvt_weightT_kernel<<<dim3((DM * DM) / 256, 4), 256, 0, stream>>>(wq, wk, wv, wo, w_bf);
  bias_table_kernel<<<((2 * S_ - 1) * H_ + 255) / 256, 256, 0, stream>>>(table, biasbf);
  qkv_gemm_kernel<<<dim3((B_ * S_) / 128, 3 * H_), 256, 0, stream>>>(hid_bf, w_bf,
                                                                     qbuf, kbuf, vbuf);
  flash_attn_kernel<<<dim3(S_ / 128, B_ * H_), 256, 0, stream>>>(qbuf, kbuf, vbuf,
                                                                 biasbf, abuf);
  out_gemm_kernel<<<dim3((B_ * S_) / 128, DM / 64), 256, 0, stream>>>(
      abuf, w_bf + (size_t)3 * DM * DM, out);
}
